// JambaBlock_15461882265905
// MI455X (gfx1250) — matrix operand order repaired and checked by reading, not compiled, not run on hardware
//
#include <hip/hip_runtime.h>
#include <hip/hip_bf16.h>
#include <math.h>
#include <stdint.h>

// ---------------- problem constants (match reference) ----------------
#define DIM      1024
#define D_STATE  16
#define D_CONV   4
#define D_INNER  2048
#define DT_RANK  64
#define N_EXP    8
#define HID      2048
#define BSZ      2
#define TLEN     2048
#define NTOK     (BSZ * TLEN)             // 4096 tokens
#define PROJ_W   (DT_RANK + 2 * D_STATE)  // 96
#define EPSF     1e-5f

// tiled GEMM macro-tile
#define BM       64
#define BN       128
#define LDS_STR  36     // padded LDS row stride (floats): 36*r mod 64 distinct for r=0..15

// ---------------- WMMA fragment types (gfx1250, wave32) ----------------
typedef __attribute__((ext_vector_type(16))) _Float16 v16h;
typedef __attribute__((ext_vector_type(8)))  float    v8f;

__device__ __forceinline__ float fast_rcp(float x) { return __builtin_amdgcn_rcpf(x); }
__device__ __forceinline__ float silu_f(float x) { return x * fast_rcp(1.f + __expf(-x)); }
__device__ __forceinline__ float softplus_f(float x) { return x > 20.f ? x : __logf(1.f + __expf(x)); }

__device__ __forceinline__ v16h pack16(float4 a, float4 b, float4 c, float4 d) {
  v16h f;
  f[0]  = (_Float16)a.x; f[1]  = (_Float16)a.y; f[2]  = (_Float16)a.z; f[3]  = (_Float16)a.w;
  f[4]  = (_Float16)b.x; f[5]  = (_Float16)b.y; f[6]  = (_Float16)b.z; f[7]  = (_Float16)b.w;
  f[8]  = (_Float16)c.x; f[9]  = (_Float16)c.y; f[10] = (_Float16)c.z; f[11] = (_Float16)c.w;
  f[12] = (_Float16)d.x; f[13] = (_Float16)d.y; f[14] = (_Float16)d.z; f[15] = (_Float16)d.w;
  return f;
}

// A fragment (16x32 f16, ISA 7.12.2): half 0 -> K {0..7,16..23}, half 1 -> K {8..15,24..31}.
__device__ __forceinline__ v16h load_a_frag(const float* arow, int k0, int kh) {
  const float4* p0 = (const float4*)(arow + k0 + kh * 8);
  const float4* p1 = (const float4*)(arow + k0 + 16 + kh * 8);
  return pack16(p0[0], p0[1], p1[0], p1[1]);
}
// B fragment (32x16 f16): lane n = lane&15; half 0 -> K {0..7,16..23}, half 1 -> K {8..15,24..31}.
__device__ __forceinline__ v16h load_b_frag(const float* wrow, int k0, int kh) {
  const float4* q0 = (const float4*)(wrow + k0 + kh * 8);
  const float4* q1 = (const float4*)(wrow + k0 + 16 + kh * 8);
  return pack16(q0[0], q0[1], q1[0], q1[1]);
}

// ---------------- CDNA5 async global->LDS staging (ASYNCcnt tracked) ----------------
__device__ __forceinline__ unsigned lds_off_u32(const void* p) {
  // generic -> LDS addrspace; ptrtoint of an AS(3) pointer yields the 32-bit LDS byte offset
  return (unsigned)(uintptr_t)(__attribute__((address_space(3))) const char*)p;
}
__device__ __forceinline__ void async_copy_b128(float* lds_dst, const float* gsrc) {
  asm volatile("global_load_async_to_lds_b128 %0, %1, off"
               :: "v"(lds_off_u32(lds_dst)), "v"(gsrc) : "memory");
}
__device__ __forceinline__ void async_wait_all() {
  asm volatile("s_wait_asynccnt 0x0" ::: "memory");
}

// stage a BM x 32 fp32 A panel into LDS (padded rows); 512 b128 transfers / 256 threads
__device__ __forceinline__ void stage_a_async(const float* __restrict__ A, int lda,
                                              int rowBase, int k0, float* sbuf) {
  const int tid = threadIdx.x;
#pragma unroll
  for (int i = 0; i < 2; ++i) {
    const int idx = tid + 256 * i;    // 0..511
    const int row = idx >> 3;         // 0..63
    const int c4  = idx & 7;          // float4 column within 32-wide K chunk
    async_copy_b128(sbuf + row * LDS_STR + c4 * 4,
                    A + (size_t)(rowBase + row) * lda + k0 + c4 * 4);
  }
}

// ---------------- tiled WMMA GEMM: C[M,N] = epi(A[M,K] @ W[N,K]^T) ----------------
// grid = (N/BN, M/BM), 256 threads (8 waves as 2x4), each wave owns a 32x32 tile.
// EPI 0: store; 1: softplus(acc+bias[n]); 2: acc+resid -> C and C2;
// EPI 3: C += acc (non-atomic: one lane per element, expert launches serialized).
template <int EPI>
__global__ __launch_bounds__(256) void gemm_tiled_wmma(
    const float* __restrict__ A, int lda,
    const float* __restrict__ W, int ldw,
    float* __restrict__ C, int ldc, int K,
    const float* __restrict__ bias,
    const float* __restrict__ resid,
    float* __restrict__ C2) {
  __shared__ float sA[2][BM * LDS_STR];
  const int lane = threadIdx.x & 31;
  const int wv   = threadIdx.x >> 5;        // 0..7
  const int wy = wv >> 2, wx = wv & 3;      // 2 x 4 wave grid
  const int kh = lane >> 4, lidx = lane & 15;
  const int rowBase = blockIdx.y * BM;
  const int colBase = blockIdx.x * BN;

  stage_a_async(A, lda, rowBase, 0, sA[0]);
  async_wait_all();
  __syncthreads();

  const float* w0row = W + (size_t)(colBase + wx * 32 + lidx) * ldw;
  const float* w1row = W + (size_t)(colBase + wx * 32 + 16 + lidx) * ldw;

  v8f acc[2][2] = {};
  const int KT = K >> 5;
  for (int kt = 0; kt < KT; ++kt) {
    if (kt + 1 < KT) {
      stage_a_async(A, lda, rowBase, (kt + 1) << 5, sA[(kt + 1) & 1]);
      __builtin_prefetch(w0row + ((kt + 1) << 5) + kh * 16, 0, 1);  // global_prefetch_b8
      __builtin_prefetch(w1row + ((kt + 1) << 5) + kh * 16, 0, 1);
    }
    const float* sbuf = sA[kt & 1];
    const int k0 = kt << 5;

    v16h a0 = load_a_frag(sbuf + (wy * 32 + lidx) * LDS_STR, 0, kh);       // ds_load_b128
    v16h a1 = load_a_frag(sbuf + (wy * 32 + 16 + lidx) * LDS_STR, 0, kh);
    v16h b0 = load_b_frag(w0row, k0, kh);
    v16h b1 = load_b_frag(w1row, k0, kh);
    acc[0][0] = __builtin_amdgcn_wmma_f32_16x16x32_f16(false, a0, false, b0, (short)0, acc[0][0], false, false);
    acc[0][1] = __builtin_amdgcn_wmma_f32_16x16x32_f16(false, a0, false, b1, (short)0, acc[0][1], false, false);
    acc[1][0] = __builtin_amdgcn_wmma_f32_16x16x32_f16(false, a1, false, b0, (short)0, acc[1][0], false, false);
    acc[1][1] = __builtin_amdgcn_wmma_f32_16x16x32_f16(false, a1, false, b1, (short)0, acc[1][1], false, false);

    async_wait_all();   // next-stage copies complete (own wave) ...
    __syncthreads();    // ... and all waves' copies complete
  }

#pragma unroll
  for (int mt = 0; mt < 2; ++mt)
#pragma unroll
    for (int nt = 0; nt < 2; ++nt) {
      const int n = colBase + wx * 32 + nt * 16 + lidx;
#pragma unroll
      for (int r = 0; r < 8; ++r) {
        const int m = rowBase + wy * 32 + mt * 16 + r + kh * 8;
        float v = acc[mt][nt][r];
        if (EPI == 1) v = softplus_f(v + bias[n]);
        if (EPI == 2) v += resid[(size_t)m * ldc + n];
        if (EPI == 3) v += C[(size_t)m * ldc + n];
        C[(size_t)m * ldc + n] = v;
        if (EPI == 2) C2[(size_t)m * ldc + n] = v;
      }
    }
}

// ---------------- tiled fused MoE gate/up: H = routew * silu(A@Wg^T) * (A@Wu^T) ----------------
// Same macro-tile structure; dual B streams -> 8 WMMAs per K-step per wave.
__global__ __launch_bounds__(256) void gemm_gateup_tiled(
    const float* __restrict__ A,       // (NTOK, DIM)
    const float* __restrict__ Wg,
    const float* __restrict__ Wu,      // (HID, DIM)
    const float* __restrict__ scale,   // routew + e, stride N_EXP
    float* __restrict__ H) {           // (NTOK, HID)
  __shared__ float sA[2][BM * LDS_STR];
  const int lane = threadIdx.x & 31;
  const int wv   = threadIdx.x >> 5;
  const int wy = wv >> 2, wx = wv & 3;
  const int kh = lane >> 4, lidx = lane & 15;
  const int rowBase = blockIdx.y * BM;
  const int colBase = blockIdx.x * BN;

  stage_a_async(A, DIM, rowBase, 0, sA[0]);
  async_wait_all();
  __syncthreads();

  const float* g0row = Wg + (size_t)(colBase + wx * 32 + lidx) * DIM;
  const float* g1row = Wg + (size_t)(colBase + wx * 32 + 16 + lidx) * DIM;
  const float* u0row = Wu + (size_t)(colBase + wx * 32 + lidx) * DIM;
  const float* u1row = Wu + (size_t)(colBase + wx * 32 + 16 + lidx) * DIM;

  v8f accg[2][2] = {}, accu[2][2] = {};
  const int KT = DIM >> 5;
  for (int kt = 0; kt < KT; ++kt) {
    if (kt + 1 < KT) stage_a_async(A, DIM, rowBase, (kt + 1) << 5, sA[(kt + 1) & 1]);
    const float* sbuf = sA[kt & 1];
    const int k0 = kt << 5;

    v16h a0 = load_a_frag(sbuf + (wy * 32 + lidx) * LDS_STR, 0, kh);
    v16h a1 = load_a_frag(sbuf + (wy * 32 + 16 + lidx) * LDS_STR, 0, kh);
    v16h bg0 = load_b_frag(g0row, k0, kh);
    v16h bg1 = load_b_frag(g1row, k0, kh);
    v16h bu0 = load_b_frag(u0row, k0, kh);
    v16h bu1 = load_b_frag(u1row, k0, kh);
    accg[0][0] = __builtin_amdgcn_wmma_f32_16x16x32_f16(false, a0, false, bg0, (short)0, accg[0][0], false, false);
    accg[0][1] = __builtin_amdgcn_wmma_f32_16x16x32_f16(false, a0, false, bg1, (short)0, accg[0][1], false, false);
    accg[1][0] = __builtin_amdgcn_wmma_f32_16x16x32_f16(false, a1, false, bg0, (short)0, accg[1][0], false, false);
    accg[1][1] = __builtin_amdgcn_wmma_f32_16x16x32_f16(false, a1, false, bg1, (short)0, accg[1][1], false, false);
    accu[0][0] = __builtin_amdgcn_wmma_f32_16x16x32_f16(false, a0, false, bu0, (short)0, accu[0][0], false, false);
    accu[0][1] = __builtin_amdgcn_wmma_f32_16x16x32_f16(false, a0, false, bu1, (short)0, accu[0][1], false, false);
    accu[1][0] = __builtin_amdgcn_wmma_f32_16x16x32_f16(false, a1, false, bu0, (short)0, accu[1][0], false, false);
    accu[1][1] = __builtin_amdgcn_wmma_f32_16x16x32_f16(false, a1, false, bu1, (short)0, accu[1][1], false, false);

    async_wait_all();
    __syncthreads();
  }

#pragma unroll
  for (int mt = 0; mt < 2; ++mt)
#pragma unroll
    for (int nt = 0; nt < 2; ++nt) {
      const int n = colBase + wx * 32 + nt * 16 + lidx;
#pragma unroll
      for (int r = 0; r < 8; ++r) {
        const int m = rowBase + wy * 32 + mt * 16 + r + kh * 8;
        const float sm = scale[(size_t)m * N_EXP];
        H[(size_t)m * HID + n] = sm * silu_f(accg[mt][nt][r]) * accu[mt][nt][r];
      }
    }
}

// ---------------- simple per-wave WMMA GEMM (for N=96 x_proj) ----------------
__global__ __launch_bounds__(256) void gemm_wave_wmma(
    const float* __restrict__ A, int lda,
    const float* __restrict__ W, int ldw,
    float* __restrict__ C, int ldc,
    int M, int N, int K) {
  const int lane = threadIdx.x & 31;
  const int wid  = (blockIdx.x * blockDim.x + threadIdx.x) >> 5;
  const int tilesN = N >> 4;
  const int tm = wid / tilesN, tn = wid - tm * tilesN;
  if (tm >= (M >> 4)) return;
  const int m0 = tm << 4, n0 = tn << 4;
  const int kh = lane >> 4, lidx = lane & 15;
  const float* arow = A + (size_t)(m0 + lidx) * lda;
  const float* wrow = W + (size_t)(n0 + lidx) * ldw;

  v8f acc = {};
  for (int k0 = 0; k0 < K; k0 += 32) {
    v16h af = load_a_frag(arow, k0, kh);
    v16h bf = load_b_frag(wrow, k0, kh);
    acc = __builtin_amdgcn_wmma_f32_16x16x32_f16(false, af, false, bf, (short)0, acc, false, false);
  }
  const int n = n0 + lidx;
#pragma unroll
  for (int r = 0; r < 8; ++r)
    C[(size_t)(m0 + r + kh * 8) * ldc + n] = acc[r];
}

// ---------------- LayerNorm (one block per token) ----------------
__global__ __launch_bounds__(256) void layernorm_kernel(
    const float* __restrict__ x, const float* __restrict__ sc,
    const float* __restrict__ bi, float* __restrict__ o) {
  const int row = blockIdx.x, tid = threadIdx.x;
  const float* xr = x + (size_t)row * DIM;
  float v[4];
#pragma unroll
  for (int i = 0; i < 4; ++i) v[i] = xr[tid + 256 * i];

  __shared__ float red[256];
  red[tid] = v[0] + v[1] + v[2] + v[3]; __syncthreads();
  for (int off = 128; off; off >>= 1) { if (tid < off) red[tid] += red[tid + off]; __syncthreads(); }
  const float mu = red[0] * (1.f / DIM);
  __syncthreads();

  float q = 0.f;
#pragma unroll
  for (int i = 0; i < 4; ++i) { float d = v[i] - mu; q += d * d; }
  red[tid] = q; __syncthreads();
  for (int off = 128; off; off >>= 1) { if (tid < off) red[tid] += red[tid + off]; __syncthreads(); }
  const float rstd = rsqrtf(red[0] * (1.f / DIM) + EPSF);

  float* orow = o + (size_t)row * DIM;
#pragma unroll
  for (int i = 0; i < 4; ++i) {
    const int c = tid + 256 * i;
    orow[c] = (v[i] - mu) * rstd * sc[c] + bi[c];
  }
}

// ---------------- depthwise causal conv (D_CONV=4) + SiLU ----------------
__global__ __launch_bounds__(256) void conv_silu_kernel(
    const float* __restrict__ xz,
    const float* __restrict__ conv_state,   // (B, D_INNER, 3)
    const float* __restrict__ conv_w,       // (D_INNER, 1, 4)
    const float* __restrict__ conv_b,
    float* __restrict__ xcs) {
  const long long idx = (long long)blockIdx.x * blockDim.x + threadIdx.x;
  if (idx >= (long long)NTOK * D_INNER) return;
  const int d  = (int)(idx % D_INNER);
  const int bt = (int)(idx / D_INNER);
  const int t  = bt % TLEN;
  const int b  = bt / TLEN;

  float acc = conv_b[d];
#pragma unroll
  for (int j = 0; j < D_CONV; ++j) {
    const int ti = t - (D_CONV - 1) + j;
    const float xv = (ti >= 0)
        ? xz[((size_t)b * TLEN + ti) * (2 * D_INNER) + d]
        : conv_state[((size_t)b * D_INNER + d) * (D_CONV - 1) + (D_CONV - 1 + ti)];
    acc += conv_w[d * D_CONV + j] * xv;
  }
  xcs[(size_t)bt * D_INNER + d] = silu_f(acc);
}

// ---------------- selective scan: lane per (channel,state), shfl-reduce over states ----------------
__global__ __launch_bounds__(256) void ssm_scan_kernel(
    const float* __restrict__ dt,    // (NTOK, D_INNER)
    const float* __restrict__ proj,  // (NTOK, 96): [dt_raw | B | C]
    const float* __restrict__ xcs,   // (NTOK, D_INNER)
    const float* __restrict__ xz,    // z = cols [D_INNER, 2*D_INNER)
    const float* __restrict__ A_log, // (D_INNER, 16)
    const float* __restrict__ Dp,
    const float* __restrict__ ssm0,  // (B, D_INNER, 16)
    float* __restrict__ y) {         // (NTOK, D_INNER), gated by silu(z)
  const int g  = blockIdx.x * 256 + threadIdx.x;
  const int s  = g & 15;
  const int ch = g >> 4;                 // 0 .. B*D_INNER-1
  const int d  = ch & (D_INNER - 1);
  const int b  = ch >> 11;               // D_INNER == 2048

  float h = ssm0[((size_t)b * D_INNER + d) * D_STATE + s];
  const float Aval = -__expf(A_log[(size_t)d * D_STATE + s]);
  const float Dpd  = Dp[d];

  for (int t = 0; t < TLEN; ++t) {
    const size_t tok = (size_t)b * TLEN + t;
    const float dtv = dt[tok * D_INNER + d];
    const float xv  = xcs[tok * D_INNER + d];
    const float Bv  = proj[tok * PROJ_W + DT_RANK + s];
    const float Cv  = proj[tok * PROJ_W + DT_RANK + D_STATE + s];
    h = __expf(dtv * Aval) * h + (dtv * Bv) * xv;
    float accv = h * Cv;
    accv += __shfl_xor(accv, 8, 16);
    accv += __shfl_xor(accv, 4, 16);
    accv += __shfl_xor(accv, 2, 16);
    accv += __shfl_xor(accv, 1, 16);
    if (s == 0) {
      const float zv = xz[tok * (2 * D_INNER) + D_INNER + d];
      y[tok * D_INNER + d] = (accv + Dpd * xv) * silu_f(zv);
    }
  }
}

// ---------------- router logits (one wave per token) ----------------
__global__ __launch_bounds__(256) void router_kernel(
    const float* __restrict__ xn, const float* __restrict__ rw,
    float* __restrict__ logits) {
  const int lane = threadIdx.x & 31;
  const int tok  = blockIdx.x * 8 + (threadIdx.x >> 5);
  const float* xr = xn + (size_t)tok * DIM;
#pragma unroll
  for (int e = 0; e < N_EXP; ++e) {
    float p = 0.f;
    for (int k = lane; k < DIM; k += 32) p += xr[k] * rw[(size_t)e * DIM + k];
    for (int off = 16; off; off >>= 1) p += __shfl_xor(p, off, 32);
    if (lane == 0) logits[(size_t)tok * N_EXP + e] = p;
  }
}

// ---------------- softmax + top-2 routing weights ----------------
__global__ __launch_bounds__(256) void topk_kernel(
    const float* __restrict__ logits, float* __restrict__ routew) {
  const int t = blockIdx.x * 256 + threadIdx.x;
  if (t >= NTOK) return;
  float l[N_EXP], mx = -1e30f;
#pragma unroll
  for (int e = 0; e < N_EXP; ++e) { l[e] = logits[(size_t)t * N_EXP + e]; mx = fmaxf(mx, l[e]); }
  float p[N_EXP], ps = 0.f;
#pragma unroll
  for (int e = 0; e < N_EXP; ++e) { p[e] = __expf(l[e] - mx); ps += p[e]; }
  const float rps = fast_rcp(ps);
#pragma unroll
  for (int e = 0; e < N_EXP; ++e) p[e] *= rps;

  int i1 = 0;
#pragma unroll
  for (int e = 1; e < N_EXP; ++e) if (p[e] > p[i1]) i1 = e;
  int i2 = (i1 == 0) ? 1 : 0;
#pragma unroll
  for (int e = 0; e < N_EXP; ++e) if (e != i1 && p[e] > p[i2]) i2 = e;

  const float rw = fast_rcp(p[i1] + p[i2]);
#pragma unroll
  for (int e = 0; e < N_EXP; ++e) routew[(size_t)t * N_EXP + e] = 0.f;
  routew[(size_t)t * N_EXP + i1] = p[i1] * rw;
  routew[(size_t)t * N_EXP + i2] = p[i2] * rw;
}

// ---------------- host-side orchestration ----------------
extern "C" void kernel_launch(void* const* d_in, const int* in_sizes, int n_in,
                              void* d_out, int out_size, void* d_ws, size_t ws_size,
                              hipStream_t stream) {
  (void)in_sizes; (void)n_in; (void)out_size; (void)ws_size;
  const float* x          = (const float*)d_in[0];
  const float* ssm0       = (const float*)d_in[1];
  const float* conv_state = (const float*)d_in[2];
  const float* ln1_s      = (const float*)d_in[3];
  const float* ln1_b      = (const float*)d_in[4];
  const float* ln2_s      = (const float*)d_in[5];
  const float* ln2_b      = (const float*)d_in[6];
  const float* in_proj_w  = (const float*)d_in[7];
  const float* conv_w     = (const float*)d_in[8];
  const float* conv_b     = (const float*)d_in[9];
  const float* x_proj_w   = (const float*)d_in[10];
  const float* dt_proj_w  = (const float*)d_in[11];
  const float* dt_proj_b  = (const float*)d_in[12];
  const float* A_log      = (const float*)d_in[13];
  const float* Dp         = (const float*)d_in[14];
  const float* out_proj_w = (const float*)d_in[15];
  const float* router_w   = (const float*)d_in[16];
  const float* w_gate     = (const float*)d_in[17];
  const float* w_up       = (const float*)d_in[18];
  const float* w_down     = (const float*)d_in[19];
  float* out = (float*)d_out;

  float* w = (float*)d_ws;
  size_t off = 0;
  float* xn     = w + off; off += (size_t)NTOK * DIM;
  float* xz     = w + off; off += (size_t)NTOK * 2 * D_INNER;
  float* xcs    = w + off; off += (size_t)NTOK * D_INNER;
  float* proj   = w + off; off += (size_t)NTOK * PROJ_W;
  float* dtbuf  = w + off; off += (size_t)NTOK * D_INNER;
  float* ybuf   = w + off; off += (size_t)NTOK * D_INNER;   // scan out, reused as MoE h
  float* x1     = w + off; off += (size_t)NTOK * DIM;
  float* xn2    = w + off; off += (size_t)NTOK * DIM;
  float* logits = w + off; off += (size_t)NTOK * N_EXP;
  float* routew = w + off; off += (size_t)NTOK * N_EXP;

  // 1) LN1
  layernorm_kernel<<<NTOK, 256, 0, stream>>>(x, ln1_s, ln1_b, xn);

  // 2) in_proj: (4096,1024) x (4096,1024)^T -> xz
  gemm_tiled_wmma<0><<<dim3((2 * D_INNER) / BN, NTOK / BM), 256, 0, stream>>>(
      xn, DIM, in_proj_w, DIM, xz, 2 * D_INNER, DIM, nullptr, nullptr, nullptr);

  // 3) depthwise conv + silu
  conv_silu_kernel<<<((long long)NTOK * D_INNER) / 256, 256, 0, stream>>>(
      xz, conv_state, conv_w, conv_b, xcs);

  // 4) x_proj: N=96 -> simple per-wave kernel
  {
    int waves = (NTOK / 16) * (PROJ_W / 16);
    gemm_wave_wmma<<<waves / 8, 256, 0, stream>>>(
        xcs, D_INNER, x_proj_w, D_INNER, proj, PROJ_W, NTOK, PROJ_W, D_INNER);
  }

  // 5) dt_proj + softplus: (4096, 64-slice of proj) x (2048,64)^T -> dt
  gemm_tiled_wmma<1><<<dim3(D_INNER / BN, NTOK / BM), 256, 0, stream>>>(
      proj, PROJ_W, dt_proj_w, DT_RANK, dtbuf, D_INNER, DT_RANK,
      dt_proj_b, nullptr, nullptr);

  // 6) selective scan (+ silu(z) gating); lane per (channel, state)
  ssm_scan_kernel<<<(BSZ * D_INNER * D_STATE) / 256, 256, 0, stream>>>(
      dtbuf, proj, xcs, xz, A_log, Dp, ssm0, ybuf);

  // 7) out_proj + residual: x1 = x + y @ out_proj_w^T ; d_out initialized to x1
  gemm_tiled_wmma<2><<<dim3(DIM / BN, NTOK / BM), 256, 0, stream>>>(
      ybuf, D_INNER, out_proj_w, D_INNER, x1, DIM, D_INNER,
      nullptr, x, out);

  // 8) LN2
  layernorm_kernel<<<NTOK, 256, 0, stream>>>(x1, ln2_s, ln2_b, xn2);

  // 9) router + top-2
  router_kernel<<<NTOK / 8, 256, 0, stream>>>(xn2, router_w, logits);
  topk_kernel<<<NTOK / 256, 256, 0, stream>>>(logits, routew);

  // 10) MoE: per expert, tiled fused gate/up (scaled by routing weight; row
  //     scaling commutes with the down projection), then down-proj accumulated
  //     into d_out (non-atomic: expert launches are stream-serialized and each
  //     element is owned by exactly one lane per launch).
  for (int e = 0; e < N_EXP; ++e) {
    const float* Wg = w_gate + (size_t)e * HID * DIM;
    const float* Wu = w_up   + (size_t)e * HID * DIM;
    const float* Wd = w_down + (size_t)e * DIM * HID;

    gemm_gateup_tiled<<<dim3(HID / BN, NTOK / BM), 256, 0, stream>>>(
        xn2, Wg, Wu, routew + e, ybuf);

    gemm_tiled_wmma<3><<<dim3(DIM / BN, NTOK / BM), 256, 0, stream>>>(
        ybuf, HID, Wd, HID, out, DIM, HID, nullptr, nullptr, nullptr);
  }
}
